// MultiHeadAttention_10307921510603
// MI455X (gfx1250) — matrix operand order repaired and checked by reading, not compiled, not run on hardware
//
#include <hip/hip_runtime.h>

typedef _Float16 v16h __attribute__((ext_vector_type(16)));
typedef _Float16 v8h  __attribute__((ext_vector_type(8)));
typedef float    v8f  __attribute__((ext_vector_type(8)));

#define D_MODEL 1024
#define NHEADS  16
#define E_HEAD  64
#define BATCH   4
#define SEQ     2048
#define MROWS   (BATCH*SEQ)   /* 8192 */
#define F3      (3*D_MODEL)   /* 3072 */

#define SG_VMEM_READ 0x020
#define SG_MATRIX    0x008   /* MFMA/WMMA class in the AMDGPU scheduler */

static __device__ __forceinline__ v16h pack16(v8h lo, v8h hi) {
  return __builtin_shufflevector(lo, hi, 0,1,2,3,4,5,6,7,8,9,10,11,12,13,14,15);
}

static __device__ __forceinline__ v8f wmma_f16(v16h a, v16h b, v8f c) {
  // D = A(16x32 f16) * B(32x16 f16) + C(16x16 f32)
  return __builtin_amdgcn_wmma_f32_16x16x32_f16(false, a, false, b, (short)0, c, false, false);
}

// ---------------------------------------------------------------------------
// Kernel 0: fp32 -> fp16 cast
// ---------------------------------------------------------------------------
__global__ __launch_bounds__(256) void cvt_f32_f16(const float* __restrict__ s,
                                                   _Float16* __restrict__ d, int n) {
  int i = blockIdx.x * 256 + threadIdx.x;
  if (i < n) d[i] = (_Float16)s[i];
}

// ---------------------------------------------------------------------------
// epilogue for one 16-row group of a QKV tile: bias + RoPE + scatter
// ---------------------------------------------------------------------------
static __device__ __forceinline__ void qkv_epilogue(
    v8f acc0, v8f acc1, v8f acc2, v8f acc3,
    int mbase, int f0, int ln, int hf,
    const float* __restrict__ bias,
    _Float16* __restrict__ Qh, _Float16* __restrict__ Kh, _Float16* __restrict__ Vt)
{
  const float bv0 = bias[f0 + ln];
  const float bv1 = bias[f0 + 16 + ln];
  const float bv2 = bias[f0 + 32 + ln];
  const float bv3 = bias[f0 + 48 + ln];
  #pragma unroll
  for (int r = 0; r < 8; ++r) { acc0[r] += bv0; acc1[r] += bv1; acc2[r] += bv2; acc3[r] += bv3; }

  const int kind = f0 >> 10;            // 0=q 1=k 2=v
  const int h    = (f0 & 1023) >> 6;    // head

  if (kind < 2) {
    // RoPE: tile columns c and c+32 live in (acc0,acc2) and (acc1,acc3) at the
    // same lane/register. freq index = c (c<32), position = row % SEQ.
    const float lf  = -0.28782313662425572f; // -ln(10000)/32
    const float fr0 = __expf((float)ln * lf);
    const float fr1 = __expf((float)(16 + ln) * lf);
    #pragma unroll
    for (int r = 0; r < 8; ++r) {
      const int   row = mbase + r + 8 * hf;
      const float pos = (float)(row & (SEQ - 1));
      float c0, s0, c1, s1;
      __sincosf(pos * fr0, &s0, &c0);
      __sincosf(pos * fr1, &s1, &c1);
      float t1 = acc0[r], t2 = acc2[r];
      acc0[r] = t1 * c0 - t2 * s0;
      acc2[r] = t1 * s0 + t2 * c0;
      t1 = acc1[r]; t2 = acc3[r];
      acc1[r] = t1 * c1 - t2 * s1;
      acc3[r] = t1 * s1 + t2 * c1;
    }
  }

  const float qs = (kind == 0) ? 0.125f : 1.0f;   // fold 1/sqrt(E_HEAD) into Q
  #pragma unroll
  for (int r = 0; r < 8; ++r) {
    const int row = mbase + r + 8 * hf;
    const int b   = row >> 11;
    const int l   = row & (SEQ - 1);
    const size_t bh = (size_t)b * NHEADS + h;
    if (kind == 0) {
      _Float16* d = Qh + (bh * SEQ + l) * E_HEAD;
      d[ln]      = (_Float16)(acc0[r] * qs);
      d[16 + ln] = (_Float16)(acc1[r] * qs);
      d[32 + ln] = (_Float16)(acc2[r] * qs);
      d[48 + ln] = (_Float16)(acc3[r] * qs);
    } else if (kind == 1) {
      _Float16* d = Kh + (bh * SEQ + l) * E_HEAD;
      d[ln]      = (_Float16)acc0[r];
      d[16 + ln] = (_Float16)acc1[r];
      d[32 + ln] = (_Float16)acc2[r];
      d[48 + ln] = (_Float16)acc3[r];
    } else {
      _Float16* d = Vt + bh * (size_t)E_HEAD * SEQ + l;
      d[(size_t)(ln) * SEQ]      = (_Float16)acc0[r];
      d[(size_t)(16 + ln) * SEQ] = (_Float16)acc1[r];
      d[(size_t)(32 + ln) * SEQ] = (_Float16)acc2[r];
      d[(size_t)(48 + ln) * SEQ] = (_Float16)acc3[r];
    }
  }
}

// ---------------------------------------------------------------------------
// Kernel 1: QKV GEMM (8192x3072 = X[8192x1024] * Wp^T) + bias + RoPE + scatter
// One wave computes a 32x64 tile (2 A frags x 4 B frags = 8 WMMAs / k-step).
// k-loop is register-rotated and the schedule is pinned with
// sched_group_barrier: [12 vmem reads][8 wmma] per step, so step k's WMMAs
// overlap step k+1's loads (partial s_wait_loadcnt instead of 0).
// ---------------------------------------------------------------------------
__global__ __launch_bounds__(256) void qkv_rope(
    const _Float16* __restrict__ X, const _Float16* __restrict__ W,
    const float* __restrict__ bias,
    _Float16* __restrict__ Qh, _Float16* __restrict__ Kh, _Float16* __restrict__ Vt)
{
  const int lane = threadIdx.x & 31;
  const int wv   = threadIdx.x >> 5;
  const int ln   = lane & 15;
  const int hf   = lane >> 4;
  const int wt   = blockIdx.x * 8 + wv;           // 0..12287
  const int ft   = wt % (F3 / 64);                // 0..47
  const int mt   = wt / (F3 / 64);                // 0..255
  const int m0   = mt * 32;
  const int f0   = ft * 64;

  v8f c00 = {}, c01 = {}, c02 = {}, c03 = {};
  v8f c10 = {}, c11 = {}, c12 = {}, c13 = {};
  const _Float16* arow0 = X + (size_t)(m0 + ln) * D_MODEL + hf * 16;
  const _Float16* arow1 = arow0 + (size_t)16 * D_MODEL;
  const _Float16* brow  = W + (size_t)(f0 + ln) * D_MODEL + hf * 16;

  // prologue: fragments for k0 = 0
  v16h a0 = pack16(*(const v8h*)(arow0), *(const v8h*)(arow0 + 8));
  v16h a1 = pack16(*(const v8h*)(arow1), *(const v8h*)(arow1 + 8));
  v16h b0 = *(const v16h*)(brow);
  v16h b1 = *(const v16h*)(brow + 16 * D_MODEL);
  v16h b2 = *(const v16h*)(brow + 32 * D_MODEL);
  v16h b3 = *(const v16h*)(brow + 48 * D_MODEL);

  #pragma unroll 4
  for (int k0 = 32; k0 < D_MODEL; k0 += 32) {
    const _Float16* ap0 = arow0 + k0;
    const _Float16* ap1 = arow1 + k0;
    const _Float16* bp  = brow + k0;
    v16h an0 = pack16(*(const v8h*)ap0, *(const v8h*)(ap0 + 8));
    v16h an1 = pack16(*(const v8h*)ap1, *(const v8h*)(ap1 + 8));
    v16h bn0 = *(const v16h*)(bp);
    v16h bn1 = *(const v16h*)(bp + 16 * D_MODEL);
    v16h bn2 = *(const v16h*)(bp + 32 * D_MODEL);
    v16h bn3 = *(const v16h*)(bp + 48 * D_MODEL);
    c00 = wmma_f16(a0, b0, c00);
    c01 = wmma_f16(a0, b1, c01);
    c02 = wmma_f16(a0, b2, c02);
    c03 = wmma_f16(a0, b3, c03);
    c10 = wmma_f16(a1, b0, c10);
    c11 = wmma_f16(a1, b1, c11);
    c12 = wmma_f16(a1, b2, c12);
    c13 = wmma_f16(a1, b3, c13);
    // pin the schedule: this step's 12 loads first, then the 8 WMMAs
    // (which consume the PREVIOUS step's fragments thanks to the rotation)
    __builtin_amdgcn_sched_group_barrier(SG_VMEM_READ, 12, 0);
    __builtin_amdgcn_sched_group_barrier(SG_MATRIX,     8, 0);
    a0 = an0; a1 = an1; b0 = bn0; b1 = bn1; b2 = bn2; b3 = bn3;
  }
  // epilogue k-step
  c00 = wmma_f16(a0, b0, c00);
  c01 = wmma_f16(a0, b1, c01);
  c02 = wmma_f16(a0, b2, c02);
  c03 = wmma_f16(a0, b3, c03);
  c10 = wmma_f16(a1, b0, c10);
  c11 = wmma_f16(a1, b1, c11);
  c12 = wmma_f16(a1, b2, c12);
  c13 = wmma_f16(a1, b3, c13);

  qkv_epilogue(c00, c01, c02, c03, m0,      f0, ln, hf, bias, Qh, Kh, Vt);
  qkv_epilogue(c10, c11, c12, c13, m0 + 16, f0, ln, hf, bias, Qh, Kh, Vt);
}

// ---------------------------------------------------------------------------
// helper: load the 4 K B-fragments of one 32-wide KV tile
// ---------------------------------------------------------------------------
static __device__ __forceinline__ void loadK4(const _Float16* __restrict__ Kb,
                                              int kv0, int ln, int hf,
                                              v16h& c0, v16h& c1, v16h& c2, v16h& c3) {
  const _Float16* kp = Kb + (size_t)(kv0 + ln) * E_HEAD + hf * 16;
  c0 = *(const v16h*)kp;
  c1 = *(const v16h*)(kp + 32);
  const _Float16* kp2 = kp + (size_t)16 * E_HEAD;
  c2 = *(const v16h*)kp2;
  c3 = *(const v16h*)(kp2 + 32);
}

// ---------------------------------------------------------------------------
// Kernel 2: flash attention. Block = 128 threads (4 waves), each wave owns 16
// query rows; block covers 64 rows of one (b,h). KV tiles of 32.
// Pipelining: V frags for tile t loaded before the S-WMMAs; K frags for tile
// t+1 loaded right after the S-WMMAs; softmax + LDS transpose hide both.
// ---------------------------------------------------------------------------
__global__ __launch_bounds__(128) void flash_attn(
    const _Float16* __restrict__ Qh, const _Float16* __restrict__ Kh,
    const _Float16* __restrict__ Vt, _Float16* __restrict__ O)
{
  __shared__ __align__(32) _Float16 lds[4 * 16 * 32];
  const int lane = threadIdx.x & 31;
  const int wv   = threadIdx.x >> 5;
  const int ln   = lane & 15;
  const int hf   = lane >> 4;
  const int bh   = blockIdx.x >> 5;                  // b*16 + h
  const int q0   = (blockIdx.x & 31) * 64 + wv * 16; // query tile start

  const _Float16* Qb = Qh + (size_t)bh * SEQ * E_HEAD;
  const _Float16* Kb = Kh + (size_t)bh * SEQ * E_HEAD;
  const _Float16* Vb = Vt + (size_t)bh * E_HEAD * SEQ;

  // Q A-fragments for d=[0,32) and d=[32,64) — reused for the whole row sweep
  v16h qa0, qa1;
  {
    const _Float16* ap = Qb + (size_t)(q0 + ln) * E_HEAD + hf * 16;
    qa0 = pack16(*(const v8h*)ap,        *(const v8h*)(ap + 8));
    qa1 = pack16(*(const v8h*)(ap + 32), *(const v8h*)(ap + 40));
  }

  v8f o0 = {}, o1 = {}, o2 = {}, o3 = {};
  float mr[8], lr[8];
  #pragma unroll
  for (int r = 0; r < 8; ++r) { mr[r] = -3.0e38f; lr[r] = 0.f; }

  _Float16* pl = lds + wv * (16 * 32);

  // prologue: K fragments for kv0 = 0
  v16h k0a, k0b, k1a, k1b;
  loadK4(Kb, 0, ln, hf, k0a, k0b, k1a, k1b);

  for (int kv0 = 0; kv0 < SEQ; kv0 += 32) {
    // V fragments for this tile — needed only after softmax; issue early
    const _Float16* vp = Vb + (size_t)ln * SEQ + kv0 + hf * 16;
    v16h vb0 = *(const v16h*)(vp);
    v16h vb1 = *(const v16h*)(vp + (size_t)16 * SEQ);
    v16h vb2 = *(const v16h*)(vp + (size_t)32 * SEQ);
    v16h vb3 = *(const v16h*)(vp + (size_t)48 * SEQ);

    // S tiles: cols [kv0,kv0+16) in sa, [kv0+16,kv0+32) in sb (scale in Q)
    v8f sa = {}, sb = {};
    sa = wmma_f16(qa0, k0a, sa);
    sa = wmma_f16(qa1, k0b, sa);
    sb = wmma_f16(qa0, k1a, sb);
    sb = wmma_f16(qa1, k1b, sb);

    // prefetch next tile's K fragments (wrapped index keeps body branch-free)
    const int kvn = (kv0 + 32 < SEQ) ? (kv0 + 32) : 0;
    v16h n0a, n0b, n1a, n1b;
    loadK4(Kb, kvn, ln, hf, n0a, n0b, n1a, n1b);

    // online softmax (rows live per-register across a 16-lane half)
    float fac[8];
    #pragma unroll
    for (int r = 0; r < 8; ++r) {
      float mx = fmaxf(sa[r], sb[r]);
      mx = fmaxf(mx, __shfl_xor(mx, 1));
      mx = fmaxf(mx, __shfl_xor(mx, 2));
      mx = fmaxf(mx, __shfl_xor(mx, 4));
      mx = fmaxf(mx, __shfl_xor(mx, 8));
      const float mn = fmaxf(mr[r], mx);
      const float p0 = __expf(sa[r] - mn);
      const float p1 = __expf(sb[r] - mn);
      float rs = p0 + p1;
      rs += __shfl_xor(rs, 1);
      rs += __shfl_xor(rs, 2);
      rs += __shfl_xor(rs, 4);
      rs += __shfl_xor(rs, 8);
      fac[r] = __expf(mr[r] - mn);
      lr[r]  = lr[r] * fac[r] + rs;
      mr[r]  = mn;
      sa[r] = p0; sb[r] = p1;
    }
    #pragma unroll
    for (int r = 0; r < 8; ++r) { o0[r] *= fac[r]; o1[r] *= fac[r]; o2[r] *= fac[r]; o3[r] *= fac[r]; }

    // C-layout -> A-layout transpose of P through per-wave LDS
    #pragma unroll
    for (int r = 0; r < 8; ++r) {
      const int m = r + 8 * hf;
      pl[m * 32 + ln]      = (_Float16)sa[r];
      pl[m * 32 + 16 + ln] = (_Float16)sb[r];
    }
    asm volatile("s_wait_dscnt 0x0" ::: "memory");
    v16h pa;
    {
      const _Float16* ap2 = pl + ln * 32 + hf * 16;
      pa = pack16(*(const v8h*)ap2, *(const v8h*)(ap2 + 8));
    }
    asm volatile("" ::: "memory");

    // O += P * V   (V fragments already resident)
    o0 = wmma_f16(pa, vb0, o0);
    o1 = wmma_f16(pa, vb1, o1);
    o2 = wmma_f16(pa, vb2, o2);
    o3 = wmma_f16(pa, vb3, o3);

    k0a = n0a; k0b = n0b; k1a = n1a; k1b = n1b;
  }

  const int b = bh >> 4, h = bh & 15;
  #pragma unroll
  for (int r = 0; r < 8; ++r) {
    const int   qm  = q0 + r + 8 * hf;
    const float inv = 1.0f / lr[r];
    _Float16* d = O + ((size_t)(b * SEQ + qm)) * D_MODEL + h * E_HEAD;
    d[ln]      = (_Float16)(o0[r] * inv);
    d[16 + ln] = (_Float16)(o1[r] * inv);
    d[32 + ln] = (_Float16)(o2[r] * inv);
    d[48 + ln] = (_Float16)(o3[r] * inv);
  }
}

// ---------------------------------------------------------------------------
// Kernel 3: output projection 8192x1024 = Attn[8192x1024] * Wout^T + b, fp32
// out. Same 32x64 pipelined tile as qkv_rope.
// ---------------------------------------------------------------------------
__global__ __launch_bounds__(256) void out_proj(
    const _Float16* __restrict__ A, const _Float16* __restrict__ W,
    const float* __restrict__ bias, float* __restrict__ out)
{
  const int lane = threadIdx.x & 31;
  const int wv   = threadIdx.x >> 5;
  const int ln   = lane & 15;
  const int hf   = lane >> 4;
  const int wt   = blockIdx.x * 8 + wv;   // 0..4095
  const int nt   = wt & 15;
  const int mt   = wt >> 4;               // 0..255
  const int m0   = mt * 32, n0 = nt * 64;

  v8f c00 = {}, c01 = {}, c02 = {}, c03 = {};
  v8f c10 = {}, c11 = {}, c12 = {}, c13 = {};
  const _Float16* arow0 = A + (size_t)(m0 + ln) * D_MODEL + hf * 16;
  const _Float16* arow1 = arow0 + (size_t)16 * D_MODEL;
  const _Float16* brow  = W + (size_t)(n0 + ln) * D_MODEL + hf * 16;

  v16h a0 = pack16(*(const v8h*)(arow0), *(const v8h*)(arow0 + 8));
  v16h a1 = pack16(*(const v8h*)(arow1), *(const v8h*)(arow1 + 8));
  v16h b0 = *(const v16h*)(brow);
  v16h b1 = *(const v16h*)(brow + 16 * D_MODEL);
  v16h b2 = *(const v16h*)(brow + 32 * D_MODEL);
  v16h b3 = *(const v16h*)(brow + 48 * D_MODEL);

  #pragma unroll 4
  for (int k0 = 32; k0 < D_MODEL; k0 += 32) {
    const _Float16* ap0 = arow0 + k0;
    const _Float16* ap1 = arow1 + k0;
    const _Float16* bp  = brow + k0;
    v16h an0 = pack16(*(const v8h*)ap0, *(const v8h*)(ap0 + 8));
    v16h an1 = pack16(*(const v8h*)ap1, *(const v8h*)(ap1 + 8));
    v16h bn0 = *(const v16h*)(bp);
    v16h bn1 = *(const v16h*)(bp + 16 * D_MODEL);
    v16h bn2 = *(const v16h*)(bp + 32 * D_MODEL);
    v16h bn3 = *(const v16h*)(bp + 48 * D_MODEL);
    c00 = wmma_f16(a0, b0, c00);
    c01 = wmma_f16(a0, b1, c01);
    c02 = wmma_f16(a0, b2, c02);
    c03 = wmma_f16(a0, b3, c03);
    c10 = wmma_f16(a1, b0, c10);
    c11 = wmma_f16(a1, b1, c11);
    c12 = wmma_f16(a1, b2, c12);
    c13 = wmma_f16(a1, b3, c13);
    __builtin_amdgcn_sched_group_barrier(SG_VMEM_READ, 12, 0);
    __builtin_amdgcn_sched_group_barrier(SG_MATRIX,     8, 0);
    a0 = an0; a1 = an1; b0 = bn0; b1 = bn1; b2 = bn2; b3 = bn3;
  }
  c00 = wmma_f16(a0, b0, c00);
  c01 = wmma_f16(a0, b1, c01);
  c02 = wmma_f16(a0, b2, c02);
  c03 = wmma_f16(a0, b3, c03);
  c10 = wmma_f16(a1, b0, c10);
  c11 = wmma_f16(a1, b1, c11);
  c12 = wmma_f16(a1, b2, c12);
  c13 = wmma_f16(a1, b3, c13);

  const float bv0 = bias[n0 + ln];
  const float bv1 = bias[n0 + 16 + ln];
  const float bv2 = bias[n0 + 32 + ln];
  const float bv3 = bias[n0 + 48 + ln];
  #pragma unroll
  for (int r = 0; r < 8; ++r) {
    float* d = out + (size_t)(m0 + r + 8 * hf) * D_MODEL + n0;
    d[ln]      = c00[r] + bv0;
    d[16 + ln] = c01[r] + bv1;
    d[32 + ln] = c02[r] + bv2;
    d[48 + ln] = c03[r] + bv3;
    float* d2 = out + (size_t)(m0 + 16 + r + 8 * hf) * D_MODEL + n0;
    d2[ln]      = c10[r] + bv0;
    d2[16 + ln] = c11[r] + bv1;
    d2[32 + ln] = c12[r] + bv2;
    d2[48 + ln] = c13[r] + bv3;
  }
}

// ---------------------------------------------------------------------------
extern "C" void kernel_launch(void* const* d_in, const int* in_sizes, int n_in,
                              void* d_out, int out_size, void* d_ws, size_t ws_size,
                              hipStream_t stream)
{
  (void)in_sizes; (void)n_in; (void)out_size; (void)ws_size;
  const float* x  = (const float*)d_in[0];
  const float* Wp = (const float*)d_in[1];
  const float* bp = (const float*)d_in[2];
  const float* Wo = (const float*)d_in[3];
  const float* bo = (const float*)d_in[4];

  char* ws = (char*)d_ws;
  size_t off = 0;
  _Float16* xh    = (_Float16*)(ws + off); off += (size_t)MROWS * D_MODEL * 2;   // 16 MB
  _Float16* wqkvh = (_Float16*)(ws + off); off += (size_t)F3 * D_MODEL * 2;      //  6 MB
  _Float16* wouth = (_Float16*)(ws + off); off += (size_t)D_MODEL * D_MODEL * 2; //  2 MB
  _Float16* Qh    = (_Float16*)(ws + off); off += (size_t)MROWS * D_MODEL * 2;   // 16 MB
  _Float16* Kh    = (_Float16*)(ws + off); off += (size_t)MROWS * D_MODEL * 2;   // 16 MB
  _Float16* Vt    = (_Float16*)(ws + off); off += (size_t)MROWS * D_MODEL * 2;   // 16 MB
  _Float16* attnh = (_Float16*)(ws + off); off += (size_t)MROWS * D_MODEL * 2;   // 16 MB

  const int nX  = MROWS * D_MODEL;
  const int nWp = F3 * D_MODEL;
  const int nWo = D_MODEL * D_MODEL;
  cvt_f32_f16<<<(nX  + 255) / 256, 256, 0, stream>>>(x,  xh,    nX);
  cvt_f32_f16<<<(nWp + 255) / 256, 256, 0, stream>>>(Wp, wqkvh, nWp);
  cvt_f32_f16<<<(nWo + 255) / 256, 256, 0, stream>>>(Wo, wouth, nWo);

  qkv_rope<<<(MROWS / 32) * (F3 / 64) / 8, 256, 0, stream>>>(xh, wqkvh, bp, Qh, Kh, Vt);
  flash_attn<<<BATCH * NHEADS * (SEQ / 64), 128, 0, stream>>>(Qh, Kh, Vt, attnh);
  out_proj<<<(MROWS / 32) * (D_MODEL / 64) / 8, 256, 0, stream>>>(attnh, wouth, bo, (float*)d_out);
}
